// RotaryAttention_35433480192737
// MI455X (gfx1250) — matrix operand order repaired and checked by reading, not compiled, not run on hardware
//
#include <hip/hip_runtime.h>
#include <hip/hip_bf16.h>
#include <math.h>

typedef __bf16 bf16_t;
typedef __attribute__((ext_vector_type(16))) __bf16 v16bf;
typedef __attribute__((ext_vector_type(8)))  float   v8f;
typedef __attribute__((ext_vector_type(8)))  __bf16  bf16x8;
typedef __attribute__((ext_vector_type(4)))  unsigned v4u;
typedef __attribute__((ext_vector_type(4)))  int      v4i;
typedef __attribute__((ext_vector_type(8)))  int      v8i;

union Frag16 { v16bf v; bf16x8 h[2]; };

// Problem constants
constexpr int kBatch = 8;
constexpr int kSeq   = 1024;
constexpr int kDim   = 2048;
constexpr int kHeads = 16;
constexpr int kGrp   = 2;
constexpr int kHD    = 128;
constexpr int kM     = kBatch * kSeq;      // 8192 rows
constexpr int kNKV   = kGrp * kHD;         // 256

#if defined(__has_builtin)
#  if __has_builtin(__builtin_amdgcn_tensor_load_to_lds)
#    define HAVE_TDM 1
#  endif
#endif
#ifndef HAVE_TDM
#  define HAVE_TDM 0
#endif

__device__ __forceinline__ void wave_lds_sync() {
    asm volatile("s_wait_dscnt 0" ::: "memory");
}

#if HAVE_TDM
// ---------------------------------------------------------------------------
// Tensor Data Mover: async 2D tile load (global -> LDS), 2-byte elements,
// with LDS row padding of 16B per 128B row (tile_dim0 = 64 elems = 128B).
// D# layout per CDNA5 ISA 8.3/8.4. Issued per-wave, tracked by TENSORcnt.
// ---------------------------------------------------------------------------
__device__ __forceinline__ void tdm_load_tile_2d(unsigned lds_addr,
                                                 const void* gptr,
                                                 unsigned tile0, unsigned tile1,
                                                 unsigned dim0, unsigned dim1,
                                                 unsigned stride0) {
    unsigned long long ga = (unsigned long long)(uintptr_t)gptr;
    v4u g0 = { 1u,                                   // count=1, user desc
               lds_addr,                             // lds_addr (bytes)
               (unsigned)ga,                         // global_addr[31:0]
               (unsigned)((ga >> 32) & 0x01FFFFFFu) | (2u << 30) };  // +type=2
    // pad_enable=1, pad_interval=4 (32 DWORD = 128B), pad_amount=3 (4 DWORD = 16B)
    v8i g1 = { (int)((1u << 16) | (1u << 20) | (4u << 22) | (3u << 25)),
               (int)((dim0 & 0xFFFFu) << 16),
               (int)((dim0 >> 16) | ((dim1 & 0xFFFFu) << 16)),
               (int)((dim1 >> 16) | (tile0 << 16)),
               (int)(tile1),                          // tile_dim1 | tile_dim2=0
               (int)stride0,                          // tensor_dim0_stride lo
               0, 0 };
    v4i z4 = {0, 0, 0, 0};
#if __clang_major__ >= 23
    v8i z8 = {0, 0, 0, 0, 0, 0, 0, 0};
    __builtin_amdgcn_tensor_load_to_lds(g0, g1, z4, z4, z8, 0);
#else
    __builtin_amdgcn_tensor_load_to_lds(g0, g1, z4, z4, 0);
#endif
}
#endif

// ---------------------------------------------------------------------------
// fp32 -> bf16 conversion (grid-stride)
// ---------------------------------------------------------------------------
__global__ void cvt_bf16_kernel(const float* __restrict__ in,
                                bf16_t* __restrict__ out, size_t n) {
    size_t i = (size_t)blockIdx.x * blockDim.x + threadIdx.x;
    size_t stride = (size_t)gridDim.x * blockDim.x;
    for (; i < n; i += stride) out[i] = (bf16_t)in[i];
}

// ---------------------------------------------------------------------------
// fp32 [K][N] -> bf16 transposed [N][K] (tiled through LDS)
// ---------------------------------------------------------------------------
__global__ __launch_bounds__(256)
void cvt_transpose_kernel(const float* __restrict__ in, bf16_t* __restrict__ out,
                          int Kd, int Nd) {
    __shared__ float tile[32][33];
    const int k0 = blockIdx.y * 32, n0 = blockIdx.x * 32;
    const int tx = threadIdx.x, ty = threadIdx.y;
#pragma unroll
    for (int i = ty; i < 32; i += 8)
        tile[i][tx] = in[(size_t)(k0 + i) * Nd + n0 + tx];
    __syncthreads();
#pragma unroll
    for (int i = ty; i < 32; i += 8)
        out[(size_t)(n0 + i) * Kd + k0 + tx] = (bf16_t)tile[tx][i];
}

// ---------------------------------------------------------------------------
// bf16 GEMM: C[M,N] = A[M,K] @ Bt[N,K]^T + bias, fp32 accumulate via WMMA.
// Block 256 threads (8 waves). Tile BM=128 x BN=128, BK=64, double-buffered
// LDS staged by the Tensor Data Mover (wave 0 issues, TENSORcnt synced).
// Waves: 4 in M x 2 in N; each wave computes 32x64 = 2x4 WMMA frags.
// LDS rows padded to 72 elems (144B) via TDM pad for bank spread.
// ---------------------------------------------------------------------------
template <bool OUT_BF16>
__global__ __launch_bounds__(256)
void gemm_bf16_kernel(const bf16_t* __restrict__ A,
                      const bf16_t* __restrict__ Bt,   // [N][K]
                      const float* __restrict__ bias,
                      float* __restrict__ Cf, bf16_t* __restrict__ Cb,
                      int M, int N, int K) {
    constexpr int BM = 128, BN = 128, BK = 64, LDT = BK + 8;  // 144B rows
    __shared__ __align__(16) bf16_t As[2][BM][LDT];   // 2 x 18KB
    __shared__ __align__(16) bf16_t Bs[2][BN][LDT];   // 2 x 18KB

    const int tid  = threadIdx.x;
    const int wave = tid >> 5;
    const int lane = tid & 31;
    const int m0 = blockIdx.y * BM;
    const int n0 = blockIdx.x * BN;
    const int wm = (wave >> 1) * 32;
    const int wn = (wave & 1) * 64;
    const int lr = lane & 15;
    const int lh = lane >> 4;

    v8f acc[2][4] = {};

    auto stage = [&](int b, int k0) {
#if HAVE_TDM
        if (wave == 0) {
            tdm_load_tile_2d((unsigned)(uintptr_t)&As[b][0][0],
                             A + (size_t)m0 * K + k0,
                             BK, BM, (unsigned)K, (unsigned)M, (unsigned)K);
            tdm_load_tile_2d((unsigned)(uintptr_t)&Bs[b][0][0],
                             Bt + (size_t)n0 * K + k0,
                             BK, BN, (unsigned)K, (unsigned)N, (unsigned)K);
        }
#else
#pragma unroll
        for (int it = 0; it < 4; ++it) {
            const int id = tid + it * 256;           // 1024 chunks of 8
            const int r = id >> 3, c = (id & 7) * 8;
            *(bf16x8*)&As[b][r][c] =
                *(const bf16x8*)(A + (size_t)(m0 + r) * K + k0 + c);
            *(bf16x8*)&Bs[b][r][c] =
                *(const bf16x8*)(Bt + (size_t)(n0 + r) * K + k0 + c);
        }
#endif
    };

    stage(0, 0);
#if HAVE_TDM
    if (wave == 0) __builtin_amdgcn_s_wait_tensorcnt(0);
#endif
    __syncthreads();

    int buf = 0;
    for (int k0 = 0; k0 < K; k0 += BK) {
        if (k0 + BK < K) stage(buf ^ 1, k0 + BK);   // async prefetch next tile

#pragma unroll
        for (int kk = 0; kk < BK; kk += 32) {
            Frag16 af[2], bfr[4];
#pragma unroll
            for (int mi = 0; mi < 2; ++mi) {
                const int row = wm + mi * 16 + lr;
                const int kb  = kk + lh * 8;
                af[mi].h[0] = *(const bf16x8*)&As[buf][row][kb];
                af[mi].h[1] = *(const bf16x8*)&As[buf][row][kb + 16];
            }
#pragma unroll
            for (int ni = 0; ni < 4; ++ni) {
                const int col = wn + ni * 16 + lr;
                const int kb  = kk + lh * 8;
                bfr[ni].h[0] = *(const bf16x8*)&Bs[buf][col][kb];
                bfr[ni].h[1] = *(const bf16x8*)&Bs[buf][col][kb + 16];
            }
#pragma unroll
            for (int mi = 0; mi < 2; ++mi)
#pragma unroll
                for (int ni = 0; ni < 4; ++ni)
                    acc[mi][ni] = __builtin_amdgcn_wmma_f32_16x16x32_bf16(
                        false, af[mi].v, false, bfr[ni].v,
                        (short)0, acc[mi][ni], false, false);
        }

#if HAVE_TDM
        if (wave == 0) __builtin_amdgcn_s_wait_tensorcnt(0);
#endif
        __syncthreads();
        buf ^= 1;
    }

    // Epilogue: lane holds col (lane&15), rows i + 8*(lane>>4)
#pragma unroll
    for (int ni = 0; ni < 4; ++ni) {
        const int n = n0 + wn + ni * 16 + lr;
        const float bv = bias[n];
#pragma unroll
        for (int mi = 0; mi < 2; ++mi) {
#pragma unroll
            for (int i = 0; i < 8; ++i) {
                const int m = m0 + wm + mi * 16 + i + lh * 8;
                const float v = acc[mi][ni][i] + bv;
                if (OUT_BF16) Cb[(size_t)m * N + n] = (bf16_t)v;
                else          Cf[(size_t)m * N + n] = v;
            }
        }
    }
}

// ---------------------------------------------------------------------------
// RoPE in place on bf16 (rows = B*S, layout [bs][h][hd])
// ---------------------------------------------------------------------------
__global__ void rope_kernel(bf16_t* __restrict__ X, int nh, size_t total) {
    size_t idx = (size_t)blockIdx.x * blockDim.x + threadIdx.x;
    if (idx >= total) return;
    const int i  = (int)(idx % 64);
    const int h  = (int)((idx / 64) % nh);
    const size_t bs = idx / (64 * (size_t)nh);
    const int s  = (int)(bs % kSeq);
    const float inv_freq = __powf(10000.f, -(float)i / 64.f);
    float sn, cs;
    __sincosf((float)s * inv_freq, &sn, &cs);
    const size_t base = (bs * nh + h) * kHD;
    const float x1 = (float)X[base + i];
    const float x2 = (float)X[base + i + 64];
    X[base + i]      = (bf16_t)(x1 * cs - x2 * sn);
    X[base + i + 64] = (bf16_t)(x2 * cs + x1 * sn);
}

// ---------------------------------------------------------------------------
// GQA flash attention. Block = 128 threads (4 waves), 64 query rows/block,
// key tiles of 32. grid = (S/64, B*H). Each wave owns 16 query rows.
// ---------------------------------------------------------------------------
__global__ __launch_bounds__(128)
void flash_attn_kernel(const bf16_t* __restrict__ Q,
                       const bf16_t* __restrict__ Kg,
                       const bf16_t* __restrict__ Vg,
                       bf16_t* __restrict__ O) {
    constexpr int QB = 64, KTILE = 32;
    const float scale = 0.088388347648318f;       // 1/sqrt(128)
    const int wave = threadIdx.x >> 5;
    const int lane = threadIdx.x & 31;
    const int lr = lane & 15, lh = lane >> 4;
    const int bh = blockIdx.y;
    const int b = bh / kHeads, h = bh % kHeads;
    const int g = h / (kHeads / kGrp);
    const int q0 = blockIdx.x * QB;

    __shared__ __align__(16) bf16_t Qs[QB][kHD];
    __shared__ __align__(16) bf16_t Ks[KTILE][kHD];     // [key][hd]
    __shared__ __align__(16) bf16_t Vst[kHD][KTILE];    // [hd][key]
    __shared__ float  Ss[4][16][KTILE];
    __shared__ __align__(16) bf16_t Ps[4][16][KTILE];
    __shared__ float arow[4][16], mrow[4][16], lrow[4][16];

    {
        const size_t qbase = ((size_t)(b * kSeq + q0)) * kDim + (size_t)h * kHD;
#pragma unroll
        for (int it = 0; it < 8; ++it) {
            const int id = threadIdx.x + it * 128;
            const int r = id >> 4, c = (id & 15) * 8;
            *(bf16x8*)&Qs[r][c] = *(const bf16x8*)(Q + qbase + (size_t)r * kDim + c);
        }
    }
    if (lane < 16) { mrow[wave][lane] = -1e30f; lrow[wave][lane] = 0.f; }
    __syncthreads();

    Frag16 qf[4];
#pragma unroll
    for (int c = 0; c < 4; ++c) {
        const int row = wave * 16 + lr;
        const int kb  = c * 32 + lh * 8;
        qf[c].h[0] = *(const bf16x8*)&Qs[row][kb];
        qf[c].h[1] = *(const bf16x8*)&Qs[row][kb + 16];
    }

    v8f accv[8] = {};

    for (int t0 = 0; t0 < kSeq; t0 += KTILE) {
        __syncthreads();
#pragma unroll
        for (int it = 0; it < 4; ++it) {
            const int id = threadIdx.x + it * 128;
            const int kr = id >> 4, c = (id & 15) * 8;
            const size_t src =
                (((size_t)(b * kSeq + t0 + kr)) * kGrp + g) * kHD + c;
            *(bf16x8*)&Ks[kr][c] = *(const bf16x8*)(Kg + src);
            bf16x8 vv = *(const bf16x8*)(Vg + src);
#pragma unroll
            for (int i = 0; i < 8; ++i) Vst[c + i][kr] = vv[i];
        }
        __syncthreads();

        // S = Q @ K^T
#pragma unroll
        for (int ni = 0; ni < 2; ++ni) {
            v8f sfrag = {};
#pragma unroll
            for (int c = 0; c < 4; ++c) {
                Frag16 kf;
                const int kn = ni * 16 + lr;
                const int kb = c * 32 + lh * 8;
                kf.h[0] = *(const bf16x8*)&Ks[kn][kb];
                kf.h[1] = *(const bf16x8*)&Ks[kn][kb + 16];
                sfrag = __builtin_amdgcn_wmma_f32_16x16x32_bf16(
                    false, qf[c].v, false, kf.v, (short)0, sfrag, false, false);
            }
#pragma unroll
            for (int i = 0; i < 8; ++i)
                Ss[wave][i + lh * 8][ni * 16 + lr] = sfrag[i] * scale;
        }
        wave_lds_sync();

        // Online softmax
        {
            const int r = lr;
            const int cb = lh * 16;
            float sv[16], mloc = -1e30f;
#pragma unroll
            for (int c = 0; c < 16; ++c) {
                sv[c] = Ss[wave][r][cb + c];
                mloc = fmaxf(mloc, sv[c]);
            }
            mloc = fmaxf(mloc, __shfl_xor(mloc, 16));
            const float mold = mrow[wave][r];
            const float mnew = fmaxf(mold, mloc);
            float sum = 0.f;
#pragma unroll
            for (int c = 0; c < 16; ++c) {
                const float p = __expf(sv[c] - mnew);
                sum += p;
                Ps[wave][r][cb + c] = (bf16_t)p;
            }
            sum += __shfl_xor(sum, 16);
            if (lh == 0) {
                const float al = __expf(mold - mnew);
                arow[wave][r] = al;
                lrow[wave][r] = lrow[wave][r] * al + sum;
                mrow[wave][r] = mnew;
            }
        }
        wave_lds_sync();

        float al[8];
#pragma unroll
        for (int i = 0; i < 8; ++i) al[i] = arow[wave][i + lh * 8];
#pragma unroll
        for (int nv = 0; nv < 8; ++nv)
#pragma unroll
            for (int i = 0; i < 8; ++i) accv[nv][i] *= al[i];

        Frag16 pf;
        {
            const int row = lr, kb = lh * 8;
            pf.h[0] = *(const bf16x8*)&Ps[wave][row][kb];
            pf.h[1] = *(const bf16x8*)&Ps[wave][row][kb + 16];
        }
#pragma unroll
        for (int nv = 0; nv < 8; ++nv) {
            Frag16 vf;
            const int hd = nv * 16 + lr;
            const int kb = lh * 8;
            vf.h[0] = *(const bf16x8*)&Vst[hd][kb];
            vf.h[1] = *(const bf16x8*)&Vst[hd][kb + 16];
            accv[nv] = __builtin_amdgcn_wmma_f32_16x16x32_bf16(
                false, pf.v, false, vf.v, (short)0, accv[nv], false, false);
        }
    }

    float li[8];
#pragma unroll
    for (int i = 0; i < 8; ++i)
        li[i] = 1.f / lrow[wave][i + lh * 8];
#pragma unroll
    for (int nv = 0; nv < 8; ++nv) {
        const int col = h * kHD + nv * 16 + lr;
#pragma unroll
        for (int i = 0; i < 8; ++i) {
            const int row = q0 + wave * 16 + i + lh * 8;
            O[((size_t)(b * kSeq + row)) * kDim + col] =
                (bf16_t)(accv[nv][i] * li[i]);
        }
    }
}

// ---------------------------------------------------------------------------
// Host-side orchestration
// ---------------------------------------------------------------------------
extern "C" void kernel_launch(void* const* d_in, const int* in_sizes, int n_in,
                              void* d_out, int out_size, void* d_ws, size_t ws_size,
                              hipStream_t stream) {
    const float* x  = (const float*)d_in[0];
    const float* Wq = (const float*)d_in[1];
    const float* bq = (const float*)d_in[2];
    const float* Wk = (const float*)d_in[3];
    const float* bk = (const float*)d_in[4];
    const float* Wv = (const float*)d_in[5];
    const float* bv = (const float*)d_in[6];
    const float* Wo = (const float*)d_in[7];
    const float* bo = (const float*)d_in[8];
    float* out = (float*)d_out;

    const size_t NX  = (size_t)kM * kDim;
    const size_t NWQ = (size_t)kDim * kDim;
    const size_t NWK = (size_t)kDim * kNKV;
    const size_t NK  = (size_t)kM * kNKV;

    char* ws = (char*)d_ws;
    size_t off = 0;
    auto alloc = [&](size_t bytes) {
        void* p = ws + off;
        off += (bytes + 255) & ~(size_t)255;
        return p;
    };
    bf16_t* xbf   = (bf16_t*)alloc(NX  * 2);
    bf16_t* WqT   = (bf16_t*)alloc(NWQ * 2);   // [N=2048][K=2048]
    bf16_t* WkT   = (bf16_t*)alloc(NWK * 2);   // [N=256][K=2048]
    bf16_t* WvT   = (bf16_t*)alloc(NWK * 2);
    bf16_t* WoT   = (bf16_t*)alloc(NWQ * 2);
    bf16_t* qbf   = (bf16_t*)alloc(NX  * 2);
    bf16_t* kbf   = (bf16_t*)alloc(NK  * 2);
    bf16_t* vbf   = (bf16_t*)alloc(NK  * 2);
    bf16_t* attbf = (bf16_t*)alloc(NX  * 2);

    // 1) fp32 -> bf16 (+ weight transpose so GEMM B tiles are TDM-friendly)
    cvt_bf16_kernel<<<2048, 256, 0, stream>>>(x, xbf, NX);
    cvt_transpose_kernel<<<dim3(kDim / 32, kDim / 32), dim3(32, 8), 0, stream>>>(
        Wq, WqT, kDim, kDim);
    cvt_transpose_kernel<<<dim3(kNKV / 32, kDim / 32), dim3(32, 8), 0, stream>>>(
        Wk, WkT, kDim, kNKV);
    cvt_transpose_kernel<<<dim3(kNKV / 32, kDim / 32), dim3(32, 8), 0, stream>>>(
        Wv, WvT, kDim, kNKV);
    cvt_transpose_kernel<<<dim3(kDim / 32, kDim / 32), dim3(32, 8), 0, stream>>>(
        Wo, WoT, kDim, kDim);

    // 2) QKV projections (bf16 out)
    gemm_bf16_kernel<true><<<dim3(kDim / 128, kM / 128), 256, 0, stream>>>(
        xbf, WqT, bq, nullptr, qbf, kM, kDim, kDim);
    gemm_bf16_kernel<true><<<dim3(kNKV / 128, kM / 128), 256, 0, stream>>>(
        xbf, WkT, bk, nullptr, kbf, kM, kNKV, kDim);
    gemm_bf16_kernel<true><<<dim3(kNKV / 128, kM / 128), 256, 0, stream>>>(
        xbf, WvT, bv, nullptr, vbf, kM, kNKV, kDim);

    // 3) RoPE on q and k
    {
        const size_t tq = (size_t)kM * kHeads * 64;
        rope_kernel<<<(unsigned)((tq + 255) / 256), 256, 0, stream>>>(
            qbf, kHeads, tq);
        const size_t tk = (size_t)kM * kGrp * 64;
        rope_kernel<<<(unsigned)((tk + 255) / 256), 256, 0, stream>>>(
            kbf, kGrp, tk);
    }

    // 4) GQA flash attention
    flash_attn_kernel<<<dim3(kSeq / 64, kBatch * kHeads), 128, 0, stream>>>(
        qbf, kbf, vbf, attbf);

    // 5) Output projection (fp32 out)
    gemm_bf16_kernel<false><<<dim3(kDim / 128, kM / 128), 256, 0, stream>>>(
        attbf, WoT, bo, out, nullptr, kM, kDim, kDim);
}
